// SubwordAggregation_77283641524505
// MI455X (gfx1250) — matrix operand order repaired and checked by reading, not compiled, not run on hardware
//
#include <hip/hip_runtime.h>

// ---------------------------------------------------------------------------
// SubwordAggregation on MI455X (gfx1250, wave32, WMMA)
//   Pass 0: W  f32 -> bf16 (RNE) into scratch (1.2 MB, L2-resident)
//   Pass 1: scores s[j] = tanh(x_j W^T + b) . v + b_s  via v_wmma_f32_16x16x32_bf16
//           block = 512 threads (16 waves), owns 64 subword rows
//           (M=64/WG halves W's L2 traffic vs M=32)
//   Pass 2: per-word stable softmax over contiguous segment + weighted sum of x
// ---------------------------------------------------------------------------

#define HIDDEN      768
#define KTILE       32
#define ROWS_PER_WG 64            // 4 M-tiles of 16
#define THREADS     512           // 16 waves
#define NT_PER_WAVE 3             // 16 waves * 3 tiles * 16 cols = 768 outputs
#define MT_PER_WAVE 4
#define LDS_STRIDE  784           // 768 + 16 bf16 pad (keeps 16B alignment)

typedef __attribute__((ext_vector_type(16))) __bf16 v16bf;
typedef __attribute__((ext_vector_type(8)))  float  v8f;

union BFrag { uint4 q[2]; v16bf v; };

static __device__ __forceinline__ unsigned short f32_to_bf16_rne(float f) {
    unsigned int u = __float_as_uint(f);
    u += 0x7fffu + ((u >> 16) & 1u);          // round-to-nearest-even
    return (unsigned short)(u >> 16);
}

// ---------------------------------------------------------------------------
// Kernel 0: convert attention_proj_w (row-major [768][768] f32) to bf16
// ---------------------------------------------------------------------------
__global__ void wconvert_kernel(const float* __restrict__ W,
                                unsigned short* __restrict__ Wb, int n4) {
    int i = blockIdx.x * blockDim.x + threadIdx.x;
    if (i >= n4) return;
    float4 f = ((const float4*)W)[i];
    uint2 p;
    p.x = (unsigned)f32_to_bf16_rne(f.x) | ((unsigned)f32_to_bf16_rne(f.y) << 16);
    p.y = (unsigned)f32_to_bf16_rne(f.z) | ((unsigned)f32_to_bf16_rne(f.w) << 16);
    ((uint2*)Wb)[i] = p;
}

// ---------------------------------------------------------------------------
// Kernel 1: WMMA GEMM + tanh + scorer dot  ->  s[n_sub]
// ---------------------------------------------------------------------------
__global__ void __launch_bounds__(THREADS)
score_gemm_kernel(const float* __restrict__ X,
                  const unsigned short* __restrict__ Wb,   // bf16 W [768][768]
                  const float* __restrict__ Pb,            // proj bias [768]
                  const float* __restrict__ Sw,            // scorer w  [768]
                  const float* __restrict__ Sb,            // scorer b  [1]
                  float* __restrict__ scores,              // out s[n_sub]
                  int n_sub) {
    __shared__ unsigned short tileA[ROWS_PER_WG][LDS_STRIDE];  // bf16 X tile (~100 KB)
    __shared__ float s_score[ROWS_PER_WG];

    const int tid  = threadIdx.x;
    const int lane = tid & 31;
    const int wave = tid >> 5;
    const int row0 = blockIdx.x * ROWS_PER_WG;

    // ---- stage X tile into LDS as bf16 (cooperative, float4 loads) ----
    for (int i = tid; i < ROWS_PER_WG * (HIDDEN / 4); i += THREADS) {
        int r  = i / (HIDDEN / 4);
        int c4 = i % (HIDDEN / 4);
        int gr = row0 + r;
        float4 f = make_float4(0.f, 0.f, 0.f, 0.f);
        if (gr < n_sub) f = ((const float4*)(X + (size_t)gr * HIDDEN))[c4];
        uint2 p;
        p.x = (unsigned)f32_to_bf16_rne(f.x) | ((unsigned)f32_to_bf16_rne(f.y) << 16);
        p.y = (unsigned)f32_to_bf16_rne(f.z) | ((unsigned)f32_to_bf16_rne(f.w) << 16);
        *(uint2*)&tileA[r][c4 * 4] = p;
    }
    if (tid < ROWS_PER_WG) s_score[tid] = 0.0f;
    __syncthreads();

    // ---- WMMA main loop: 4 M-tiles x 3 N-tiles per wave, K in steps of 32 ----
    v8f acc[MT_PER_WAVE][NT_PER_WAVE] = {};
    const int arow  = lane & 15;            // A: lanes 0-15 -> M=lane
    const int kbase = (lane >> 4) * 8;      // A: hi half-lanes hold K+8..15 / +24..31
    const int ncol0 = wave * (NT_PER_WAVE * 16);

    for (int kc = 0; kc < HIDDEN; kc += KTILE) {
        // A fragments from LDS (documented 16-bit 16x32 layout)
        v16bf afrag[MT_PER_WAVE];
#pragma unroll
        for (int m = 0; m < MT_PER_WAVE; ++m) {
            BFrag u;
            u.q[0] = *(const uint4*)&tileA[m * 16 + arow][kc + kbase];
            u.q[1] = *(const uint4*)&tileA[m * 16 + arow][kc + 16 + kbase];
            afrag[m] = u.v;
        }
        const int kstart = kc + (lane >> 4) * 8;    // B: hi half-lanes hold K+8..15 / +24..31
#pragma unroll
        for (int t = 0; t < NT_PER_WAVE; ++t) {
            const int ncol = ncol0 + t * 16 + (lane & 15);
            const unsigned short* wp = Wb + (size_t)ncol * HIDDEN + kstart;
            __builtin_prefetch(wp + KTILE, 0, 1);   // next K-slice -> global_prefetch
            BFrag ub;
            ub.q[0] = *(const uint4*)wp;
            ub.q[1] = *(const uint4*)(wp + 16);
#pragma unroll
            for (int m = 0; m < MT_PER_WAVE; ++m) {
                acc[m][t] = __builtin_amdgcn_wmma_f32_16x16x32_bf16(
                    false, afrag[m], false, ub.v, (short)0, acc[m][t], false, false);
            }
        }
    }

    // ---- epilogue: tanh + scorer dot, reduce across N, accumulate per row ----
#pragma unroll
    for (int m = 0; m < MT_PER_WAVE; ++m) {
#pragma unroll
        for (int t = 0; t < NT_PER_WAVE; ++t) {
            const int col  = ncol0 + t * 16 + (lane & 15);
            const float bias = Pb[col];
            const float sc   = Sw[col];
            v8f a = acc[m][t];
#pragma unroll
            for (int r = 0; r < 8; ++r) {
                float val = tanhf(a[r] + bias) * sc;
                // sum over the 16 lanes sharing this row (D layout: N = lane&15)
                val += __shfl_xor(val, 1, 32);
                val += __shfl_xor(val, 2, 32);
                val += __shfl_xor(val, 4, 32);
                val += __shfl_xor(val, 8, 32);
                if ((lane & 15) == 0) {
                    int row = m * 16 + r + ((lane >> 4) << 3);   // M = r + 8*(lane>=16)
                    atomicAdd(&s_score[row], val);
                }
            }
        }
    }
    __syncthreads();

    if (tid < ROWS_PER_WG) {
        int gr = row0 + tid;
        if (gr < n_sub) scores[gr] = s_score[tid] + Sb[0];
    }
}

// ---------------------------------------------------------------------------
// Kernel 2: per-word segment softmax + weighted sum of subword embeddings.
//   One wave per word (segment length <= 32; here 2..7).
// ---------------------------------------------------------------------------
__global__ void __launch_bounds__(256)
pool_kernel(const float* __restrict__ X,
            const float* __restrict__ scores,
            const int*   __restrict__ map,          // [n_words][2] start,end
            float* __restrict__ out, int n_words) {
    const int w    = (int)((blockIdx.x * (unsigned)blockDim.x + threadIdx.x) >> 5);
    const int lane = threadIdx.x & 31;
    if (w >= n_words) return;

    const int start = map[2 * w];
    const int end   = map[2 * w + 1];
    const int len   = end - start;

    // stable softmax over the segment, lane j holds subword j
    float sj = (lane < len) ? scores[start + lane] : -INFINITY;
    float m = sj;
    for (int o = 16; o > 0; o >>= 1) m = fmaxf(m, __shfl_xor(m, o, 32));
    if (m == -INFINITY) m = 0.0f;                    // empty-segment guard
    float p = (lane < len) ? __expf(sj - m) : 0.0f;
    float d = p;
    for (int o = 16; o > 0; o >>= 1) d += __shfl_xor(d, o, 32);
    if (!(d > 0.0f)) d = 1.0f;
    const float wt = p / d;

    float4 acc[6];
#pragma unroll
    for (int q = 0; q < 6; ++q) acc[q] = make_float4(0.f, 0.f, 0.f, 0.f);

    for (int j = 0; j < len; ++j) {
        const float wj = __shfl(wt, j, 32);
        const float4* xr = (const float4*)(X + (size_t)(start + j) * HIDDEN);
#pragma unroll
        for (int q = 0; q < 6; ++q) {
            float4 f = xr[lane + q * 32];
            acc[q].x += wj * f.x; acc[q].y += wj * f.y;
            acc[q].z += wj * f.z; acc[q].w += wj * f.w;
        }
    }
    float4* orow = (float4*)(out + (size_t)w * HIDDEN);
#pragma unroll
    for (int q = 0; q < 6; ++q) orow[lane + q * 32] = acc[q];
}

// ---------------------------------------------------------------------------
extern "C" void kernel_launch(void* const* d_in, const int* in_sizes, int n_in,
                              void* d_out, int out_size, void* d_ws, size_t ws_size,
                              hipStream_t stream) {
    const float* X  = (const float*)d_in[0];   // [n_sub, 768]
    const float* W  = (const float*)d_in[1];   // [768, 768]
    const float* Pb = (const float*)d_in[2];   // [768]
    const float* Sw = (const float*)d_in[3];   // [768]
    const float* Sb = (const float*)d_in[4];   // [1]
    const int*   Mp = (const int*)d_in[5];     // [n_words, 2]
    float* out = (float*)d_out;

    const int n_sub   = in_sizes[0] / HIDDEN;
    const int n_words = in_sizes[5] / 2;
    const int welems  = in_sizes[1];           // 768*768

    // scratch layout: [ bf16 W : welems*2 bytes | scores : n_sub*4 bytes ]
    unsigned short* Wb = (unsigned short*)d_ws;
    float* scores = (float*)((char*)d_ws + (size_t)welems * sizeof(unsigned short));

    // Pass 0: W -> bf16
    {
        int n4 = welems / 4;
        wconvert_kernel<<<(n4 + 255) / 256, 256, 0, stream>>>(W, Wb, n4);
    }
    // Pass 1: WMMA GEMM + tanh + scorer dot -> scores
    {
        int blocks = (n_sub + ROWS_PER_WG - 1) / ROWS_PER_WG;
        score_gemm_kernel<<<blocks, THREADS, 0, stream>>>(X, Wb, Pb, Sw, Sb, scores, n_sub);
    }
    // Pass 2: segment softmax + weighted scatter-sum
    {
        long long threads = (long long)n_words * 32;
        int blocks = (int)((threads + 255) / 256);
        pool_kernel<<<blocks, 256, 0, stream>>>(X, scores, Mp, out, n_words);
    }
}
